// GRUD_55825984914082
// MI455X (gfx1250) — compile-verified, matrix operand order checked by reading, not run on hardware
//
#include <hip/hip_runtime.h>
#include <math.h>

// ---------------------------------------------------------------------------
// GRU-D on MI455X (gfx1250, wave32).
// Persistent-workgroup RNN: weights staged in LDS as bf16, GEMMs via
// v_wmma_f32_16x16x32_bf16 with interleaved accumulator chains per wave
// (shared A fragments) so WMMA RAW hazard slots are filled with useful work.
// Hidden state resident in LDS across all T=1024 steps. Next-step activations
// prefetched into the WGP cache (global_prefetch_b8) while the GEMMs run.
// gamma_x precomputed by a parallel WMMA kernel into workspace when ws_size
// allows, else computed inline per step (also via WMMA).
// ---------------------------------------------------------------------------

#define BB 256
#define TT 1024
#define DD 64
#define HH 128
#define OO 64

typedef __bf16 bf16_t;
typedef __attribute__((ext_vector_type(16))) __bf16 v16bf;
typedef __attribute__((ext_vector_type(8)))  __bf16 v8bf;
typedef __attribute__((ext_vector_type(4)))  __bf16 v4bf;
typedef __attribute__((ext_vector_type(8)))  float  v8f;

// Load one 16x32 (MxK) bf16 fragment for WMMA from row-major [rows, ldk] data.
// CDNA5 16-bit A layout: lanes 0-15 -> rows 0-15, K = {0..7},{16..23};
// lanes 16-31 -> same rows, K base +8. B fragment of row-major W[N,K] uses the
// identical mapping with "row" = output column.
__device__ __forceinline__ v16bf load_frag_k32(const bf16_t* base, int ldk, int lane) {
    const int row = lane & 15;
    const int hi  = (lane >> 4) & 1;
    const bf16_t* p = base + row * ldk + hi * 8;
    v8bf lo = *(const v8bf*)(p);
    v8bf hv = *(const v8bf*)(p + 16);
    return __builtin_shufflevector(lo, hv, 0,1,2,3,4,5,6,7,8,9,10,11,12,13,14,15);
}

#define WMMA_BF16(A, Bm, C) \
    __builtin_amdgcn_wmma_f32_16x16x32_bf16(false, (A), false, (Bm), (short)0, (C), false, false)

__device__ __forceinline__ float sigmoidf_fast(float v) {
    return 1.0f / (1.0f + __expf(-v));
}

// ---------------------------------------------------------------------------
// Kernel 1: gamma_x = exp(-relu(deltas_x @ Wgx^T + bgx)) for all B*T rows.
// One 16-row tile per block; 4 waves own one 16-col tile each. Result staged
// through LDS so the global store is coalesced b128.
// ---------------------------------------------------------------------------
__global__ void __launch_bounds__(128)
gamma_x_kernel(const float* __restrict__ deltas_x,
               const float* __restrict__ Wgx,
               const float* __restrict__ bgx,
               bf16_t* __restrict__ gamma_out)
{
    __shared__ bf16_t sA[16 * 64];
    __shared__ bf16_t sW[64 * 64];
    __shared__ bf16_t sOut[16 * 64];
    __shared__ float  sb[64];
    const int tid  = threadIdx.x;
    const int lane = tid & 31;
    const int wv   = tid >> 5;
    const long rowBase = (long)blockIdx.x * 16;   // flattened (b*T + t) tile

    for (int i = tid; i < 16 * 64; i += 128) sA[i] = (bf16_t)deltas_x[rowBase * 64 + i];
    for (int i = tid; i < 64 * 64; i += 128) sW[i] = (bf16_t)Wgx[i];
    if (tid < 64) sb[tid] = bgx[tid];
    __syncthreads();

    const int n0 = wv * 16;
    v8f acc = {};
    #pragma unroll
    for (int k0 = 0; k0 < 64; k0 += 32) {
        v16bf a = load_frag_k32(sA + k0, 64, lane);
        v16bf b = load_frag_k32(sW + n0 * 64 + k0, 64, lane);
        acc = WMMA_BF16(a, b, acc);
    }
    const int col = n0 + (lane & 15);
    const int rhi = (lane >> 4) & 1;
    #pragma unroll
    for (int r = 0; r < 8; ++r) {
        float v = acc[r] + sb[col];
        sOut[(rhi * 8 + r) * 64 + col] = (bf16_t)__expf(-fmaxf(v, 0.0f));
    }
    __syncthreads();
    // coalesced store: 128 threads x 8 contiguous bf16 (b128 each)
    {
        const v8bf* src = (const v8bf*)sOut;
        v8bf* dst = (v8bf*)(gamma_out + rowBase * 64);
        dst[tid] = src[tid];
    }
}

// ---------------------------------------------------------------------------
// Kernel 2: persistent GRU-D scan. One block of 256 threads (8 waves) owns a
// 16-row batch tile for all 1024 timesteps. W_ih/W_hh live in LDS as bf16.
// ---------------------------------------------------------------------------
__global__ void __launch_bounds__(256)
grud_kernel(const float* __restrict__ x,
            const float* __restrict__ masks,
            const float* __restrict__ deltas_x,
            const float* __restrict__ deltas_h,
            const float* __restrict__ mean_obs,
            const float* __restrict__ Wgx,
            const float* __restrict__ bgx,
            const float* __restrict__ Wgh,
            const float* __restrict__ bgh,
            const float* __restrict__ W_ih,
            const float* __restrict__ b_ih,
            const float* __restrict__ W_hh,
            const float* __restrict__ b_hh,
            const float* __restrict__ W_out,
            const float* __restrict__ b_out,
            const bf16_t* __restrict__ gamma_x,   // used when gamma_precomp==1
            int gamma_precomp,
            float* __restrict__ out)
{
    extern __shared__ __align__(16) char smem[];
    bf16_t* sWih = (bf16_t*)smem;              // [384][64]
    bf16_t* sWhh = sWih + 384 * 64;            // [384][128]
    bf16_t* sWgx = sWhh + 384 * 128;           // [64][64]   (inline-gamma path)
    bf16_t* sAx  = sWgx + 64 * 64;             // [16][64]   x_hat (bf16)
    bf16_t* sAh  = sAx + 16 * 64;              // [16][128]  h_dec (bf16)
    float*  sRZ  = (float*)(sAh + 16 * 128);   // [16][256]  r/z pre-activations
    float*  sIN  = sRZ + 16 * 256;             // [16][128]  i_n
    float*  sHN  = sIN + 16 * 128;             // [16][128]  h_n
    float*  sH   = sHN + 16 * 128;             // [16][128]  hidden state (f32)
    float*  sHD  = sH + 16 * 128;              // [16][128]  h_dec (f32)
    float*  sLast= sHD + 16 * 128;             // [16][64]   last observation
    float*  sBrz = sLast + 16 * 64;            // [256] b_ih+b_hh for r,z
    float*  sBin = sBrz + 256;                 // [128] b_ih for n
    float*  sBhn = sBin + 128;                 // [128] b_hh for n
    float*  sWg  = sBhn + 128;                 // [128] Wgh column
    float*  sBg  = sWg + 128;                  // [128] bgh
    float*  sMean= sBg + 128;                  // [64]
    float*  sBgx = sMean + 64;                 // [64]
    float*  sGX  = sRZ;                        // [16][64] gamma_x tile (aliases sRZ)

    const int tid  = threadIdx.x;
    const int lane = tid & 31;
    const int wv   = tid >> 5;
    const int rb0  = blockIdx.x * 16;

    // contiguous thread->element mappings used every step
    const int rD  = tid >> 4;            // row for 16x64 phases (4 elems/thread)
    const int d0  = (tid & 15) * 4;
    const int rH  = tid >> 4;            // row for 16x128 phases (8 elems/thread)
    const int j0  = (tid & 15) * 8;
    const long rowStride = (long)TT * 64;          // per-batch-row stride (elems)
    const long gBase = (long)(rb0 + rD) * rowStride + d0;   // this thread's lane base

    // prefetch step 0 activations while we stage weights
    __builtin_prefetch(x + gBase, 0, 0);
    __builtin_prefetch(masks + gBase, 0, 0);
    if (gamma_precomp) __builtin_prefetch(gamma_x + gBase, 0, 0);
    else               __builtin_prefetch(deltas_x + gBase, 0, 0);

    // --- one-time staging -------------------------------------------------
    for (int i = tid; i < 384 * 64;  i += 256) sWih[i] = (bf16_t)W_ih[i];
    for (int i = tid; i < 384 * 128; i += 256) sWhh[i] = (bf16_t)W_hh[i];
    if (!gamma_precomp)
        for (int i = tid; i < 64 * 64; i += 256) sWgx[i] = (bf16_t)Wgx[i];
    sBrz[tid] = b_ih[tid] + b_hh[tid];                 // tid in [0,256)
    if (tid < 128) {
        sBin[tid] = b_ih[256 + tid];
        sBhn[tid] = b_hh[256 + tid];
        sWg[tid]  = Wgh[tid];
        sBg[tid]  = bgh[tid];
    }
    if (tid < 64) { sMean[tid] = mean_obs[tid]; sBgx[tid] = bgx[tid]; }
    for (int i = tid; i < 16 * 128; i += 256) sH[i] = 0.0f;
    for (int i = tid; i < 16 * 64;  i += 256) sLast[i] = 0.0f;
    __syncthreads();

    // --- sequential scan over T ------------------------------------------
    for (int t = 0; t < TT; ++t) {
        const long gi64 = gBase + (long)t * 64;   // x/masks/gamma offset this step

        // optional inline gamma_x tile via WMMA (waves 0..3)
        if (!gamma_precomp) {
            {
                float4 dv = *(const float4*)(deltas_x + gi64);
                v4bf o; o[0]=(bf16_t)dv.x; o[1]=(bf16_t)dv.y; o[2]=(bf16_t)dv.z; o[3]=(bf16_t)dv.w;
                *(v4bf*)(sAx + rD * 64 + d0) = o;
            }
            __syncthreads();
            if (wv < 4) {
                const int n0 = wv * 16;
                v8f acc = {};
                #pragma unroll
                for (int k0 = 0; k0 < 64; k0 += 32) {
                    v16bf a = load_frag_k32(sAx + k0, 64, lane);
                    v16bf b = load_frag_k32(sWgx + n0 * 64 + k0, 64, lane);
                    acc = WMMA_BF16(a, b, acc);
                }
                const int col = n0 + (lane & 15);
                const int rhi = (lane >> 4) & 1;
                #pragma unroll
                for (int r = 0; r < 8; ++r)
                    sGX[(rhi * 8 + r) * 64 + col] =
                        __expf(-fmaxf(acc[r] + sBgx[col], 0.0f));
            }
            __syncthreads();
        }

        // phase 1: x_hat + last_obs update (4 contiguous elems / thread)
        {
            float4 xv = *(const float4*)(x     + gi64);
            float4 mv = *(const float4*)(masks + gi64);
            float4 lo = *(const float4*)(sLast + rD * 64 + d0);
            float g[4];
            if (gamma_precomp) {
                v4bf gv = *(const v4bf*)(gamma_x + gi64);
                g[0]=(float)gv[0]; g[1]=(float)gv[1]; g[2]=(float)gv[2]; g[3]=(float)gv[3];
            } else {
                g[0]=sGX[rD*64+d0+0]; g[1]=sGX[rD*64+d0+1];
                g[2]=sGX[rD*64+d0+2]; g[3]=sGX[rD*64+d0+3];
            }
            float xa[4] = {xv.x, xv.y, xv.z, xv.w};
            float ma[4] = {mv.x, mv.y, mv.z, mv.w};
            float la[4] = {lo.x, lo.y, lo.z, lo.w};
            float4 nl; v4bf ax;
            float xh0 = ma[0]*xa[0] + (1.f-ma[0])*(g[0]*la[0] + (1.f-g[0])*sMean[d0+0]);
            float xh1 = ma[1]*xa[1] + (1.f-ma[1])*(g[1]*la[1] + (1.f-g[1])*sMean[d0+1]);
            float xh2 = ma[2]*xa[2] + (1.f-ma[2])*(g[2]*la[2] + (1.f-g[2])*sMean[d0+2]);
            float xh3 = ma[3]*xa[3] + (1.f-ma[3])*(g[3]*la[3] + (1.f-g[3])*sMean[d0+3]);
            nl.x=xh0; nl.y=xh1; nl.z=xh2; nl.w=xh3;
            ax[0]=(bf16_t)xh0; ax[1]=(bf16_t)xh1; ax[2]=(bf16_t)xh2; ax[3]=(bf16_t)xh3;
            *(float4*)(sLast + rD * 64 + d0) = nl;
            *(v4bf*)(sAx + rD * 64 + d0) = ax;
        }
        // phase 2: h_dec = h * gamma_h (8 contiguous elems / thread)
        {
            float dh = deltas_h[(long)(rb0 + rH) * TT + t];
            #pragma unroll
            for (int u = 0; u < 8; ++u) {
                int j = j0 + u;
                float gh = __expf(-fmaxf(dh * sWg[j] + sBg[j], 0.0f));
                float hd = sH[rH * 128 + j] * gh;
                sHD[rH * 128 + j] = hd;
                sAh[rH * 128 + j] = (bf16_t)hd;
            }
        }
        __syncthreads();

        // prefetch step t+1 activations into the WGP cache while the GEMMs
        // run (global_prefetch_b8: no counters, latency fully hidden).
        if (t + 1 < TT) {
            const long gn = gi64 + 64;
            __builtin_prefetch(x + gn, 0, 0);
            __builtin_prefetch(masks + gn, 0, 0);
            if (gamma_precomp) __builtin_prefetch(gamma_x + gn, 0, 0);
            else               __builtin_prefetch(deltas_x + gn, 0, 0);
        }

        // phase 3: GEMMs. Each wave owns 3 output tiles which SHARE the same
        // A fragments; the 4 accumulator chains are independent, so the
        // scheduler can interleave WMMAs instead of inserting hazard NOPs.
        {
            const int c0 = wv * 16;          // r-gate tile cols
            const int c1 = (wv + 8) * 16;    // z-gate tile cols
            const int c2 = (wv + 16) * 16;   // n-gate tile cols
            v16bf ax0 = load_frag_k32(sAx + 0,  64, lane);
            v16bf ax1 = load_frag_k32(sAx + 32, 64, lane);
            v8f a0 = {}, a1 = {}, ai = {}, an = {};
            // x_hat @ W_ih^T   (k = 0..63)
            a0 = WMMA_BF16(ax0, load_frag_k32(sWih + c0 * 64 + 0,  64, lane), a0);
            a1 = WMMA_BF16(ax0, load_frag_k32(sWih + c1 * 64 + 0,  64, lane), a1);
            ai = WMMA_BF16(ax0, load_frag_k32(sWih + c2 * 64 + 0,  64, lane), ai);
            a0 = WMMA_BF16(ax1, load_frag_k32(sWih + c0 * 64 + 32, 64, lane), a0);
            a1 = WMMA_BF16(ax1, load_frag_k32(sWih + c1 * 64 + 32, 64, lane), a1);
            ai = WMMA_BF16(ax1, load_frag_k32(sWih + c2 * 64 + 32, 64, lane), ai);
            // h_dec @ W_hh^T   (k = 0..127)
            #pragma unroll
            for (int kk = 0; kk < 4; ++kk) {
                const int k0 = kk * 32;
                v16bf ah = load_frag_k32(sAh + k0, 128, lane);
                a0 = WMMA_BF16(ah, load_frag_k32(sWhh + c0 * 128 + k0, 128, lane), a0);
                a1 = WMMA_BF16(ah, load_frag_k32(sWhh + c1 * 128 + k0, 128, lane), a1);
                an = WMMA_BF16(ah, load_frag_k32(sWhh + c2 * 128 + k0, 128, lane), an);
            }
            const int colA = lane & 15;
            const int rhi  = (lane >> 4) & 1;
            #pragma unroll
            for (int r = 0; r < 8; ++r) {
                const int row = rhi * 8 + r;
                sRZ[row * 256 + c0 + colA] = a0[r] + sBrz[c0 + colA];
                sRZ[row * 256 + c1 + colA] = a1[r] + sBrz[c1 + colA];
                sIN[row * 128 + c0 + colA] = ai[r] + sBin[c0 + colA];
                sHN[row * 128 + c0 + colA] = an[r] + sBhn[c0 + colA];
            }
        }
        __syncthreads();

        // phase 4: gates + hidden update (8 contiguous elems / thread)
        {
            #pragma unroll
            for (int u = 0; u < 8; ++u) {
                int j = j0 + u;
                float rg = sigmoidf_fast(sRZ[rH * 256 + j]);
                float zg = sigmoidf_fast(sRZ[rH * 256 + 128 + j]);
                float ng = tanhf(sIN[rH * 128 + j] + rg * sHN[rH * 128 + j]);
                sH[rH * 128 + j] = (1.0f - zg) * ng + zg * sHD[rH * 128 + j];
            }
        }
        __syncthreads();
    }

    // --- final projection: out = hT @ W_out^T + b_out (tiny; scalar FMAs) --
    {
        const int r  = tid >> 4;
        const int o0 = (tid & 15) * 4;
        float acc0 = b_out[o0+0], acc1 = b_out[o0+1], acc2 = b_out[o0+2], acc3 = b_out[o0+3];
        for (int j = 0; j < 128; ++j) {
            float hv = sH[r * 128 + j];
            acc0 += hv * W_out[(o0+0) * 128 + j];
            acc1 += hv * W_out[(o0+1) * 128 + j];
            acc2 += hv * W_out[(o0+2) * 128 + j];
            acc3 += hv * W_out[(o0+3) * 128 + j];
        }
        float4 ov; ov.x=acc0; ov.y=acc1; ov.z=acc2; ov.w=acc3;
        *(float4*)(out + (long)(rb0 + r) * 64 + o0) = ov;
    }
}

// ---------------------------------------------------------------------------
extern "C" void kernel_launch(void* const* d_in, const int* in_sizes, int n_in,
                              void* d_out, int out_size, void* d_ws, size_t ws_size,
                              hipStream_t stream) {
    const float* x        = (const float*)d_in[0];
    const float* masks    = (const float*)d_in[1];
    const float* deltas_x = (const float*)d_in[2];
    const float* deltas_h = (const float*)d_in[3];
    const float* mean_obs = (const float*)d_in[4];
    const float* Wgx      = (const float*)d_in[5];
    const float* bgx      = (const float*)d_in[6];
    const float* Wgh      = (const float*)d_in[7];
    const float* bgh      = (const float*)d_in[8];
    const float* W_ih     = (const float*)d_in[9];
    const float* b_ih     = (const float*)d_in[10];
    const float* W_hh     = (const float*)d_in[11];
    const float* b_hh     = (const float*)d_in[12];
    const float* W_out    = (const float*)d_in[13];
    const float* b_out    = (const float*)d_in[14];
    float* out            = (float*)d_out;

    const size_t gamma_bytes = (size_t)BB * TT * DD * sizeof(bf16_t); // 32 MB
    const int gamma_precomp  = (ws_size >= gamma_bytes) ? 1 : 0;
    bf16_t* gamma = (bf16_t*)d_ws;

    if (gamma_precomp) {
        gamma_x_kernel<<<(BB * TT) / 16, 128, 0, stream>>>(deltas_x, Wgx, bgx, gamma);
    }

    // Dynamic LDS: bf16 weights + activations + f32 state (~214 KB < 320 KB/WGP)
    const size_t lds =
        (size_t)(384 * 64 + 384 * 128 + 64 * 64 + 16 * 64 + 16 * 128) * sizeof(bf16_t) +
        (size_t)(16 * 256 + 16 * 128 * 4 + 16 * 64 + 256 + 128 * 4 + 64 + 64) * sizeof(float);
    (void)hipFuncSetAttribute((const void*)grud_kernel,
                              hipFuncAttributeMaxDynamicSharedMemorySize, (int)lds);

    grud_kernel<<<BB / 16, 256, lds, stream>>>(
        x, masks, deltas_x, deltas_h, mean_obs, Wgx, bgx, Wgh, bgh,
        W_ih, b_ih, W_hh, b_hh, W_out, b_out, gamma, gamma_precomp, out);
}
